// CrossAttention_87909390615125
// MI455X (gfx1250) — matrix operand order repaired and checked by reading, not compiled, not run on hardware
//
#include <hip/hip_runtime.h>
#include <stdint.h>

// ---------------- problem constants ----------------
constexpr int NB = 8;      // batch
constexpr int NT = 1024;   // T == S
constexpr int NC = 512;    // channels
constexpr float SCALE = 0.04419417382415922f; // 1/sqrt(512)

// ---------------- CDNA5 WMMA types ----------------
typedef __attribute__((ext_vector_type(16))) __bf16 v16bf;
typedef __attribute__((ext_vector_type(8)))  float  v8f;

union BFrag { uint4 q[2]; v16bf v; };

__device__ __forceinline__ v8f wmma_bf16(const v16bf& a, const v16bf& b, const v8f& c) {
  return __builtin_amdgcn_wmma_f32_16x16x32_bf16(false, a, false, b, (short)0, c,
                                                 false, false);
}

__device__ __forceinline__ unsigned short f2bf(float f) {
  unsigned u = __float_as_uint(f);
  u += 0x7FFFu + ((u >> 16) & 1u);
  return (unsigned short)(u >> 16);
}

__device__ __forceinline__ v8f vzero() {
  v8f z;
#pragma unroll
  for (int e = 0; e < 8; ++e) z[e] = 0.0f;
  return z;
}

// per-row packed-P byte offset inside the LDS scoreboard row:
// half 0 keeps p[s] at byte 2*s (within [0,1024)), half 1 at byte 2*s+1024
// (within [2048,3072)) so each lane compacts only its own half-row.
__device__ __forceinline__ int pOff(int s) { return (s < 512) ? 2 * s : 2 * s + 1024; }

// CDNA5 async copy: global -> LDS, 16 bytes, tracked by ASYNCcnt.
// INST_OFFSET is added to BOTH the global and the LDS address (ISA 08 §4.4).
__device__ __forceinline__ void async_g2l_b128(unsigned lds_off, const void* gaddr) {
  asm volatile("global_load_async_to_lds_b128 %0, %1, off"
               :: "v"(lds_off), "v"(gaddr) : "memory");
}
__device__ __forceinline__ void async_g2l_b128_o16(unsigned lds_off, const void* gaddr) {
  asm volatile("global_load_async_to_lds_b128 %0, %1, off offset:16"
               :: "v"(lds_off), "v"(gaddr) : "memory");
}
__device__ __forceinline__ void wait_async0() {
  asm volatile("s_wait_asynccnt 0x0" ::: "memory");
}

// ============= kernel 1: fp32 -> bf16 (A row-major, A^T, V row-major) =========
__global__ void k_convert(const float* __restrict__ a0, const float* __restrict__ a1,
                          const float* __restrict__ a2, const float* __restrict__ v0,
                          const float* __restrict__ v1, const float* __restrict__ v2,
                          unsigned short* __restrict__ Abf,
                          unsigned short* __restrict__ Abft,
                          unsigned short* __restrict__ Vbf) {
  size_t idx = (size_t)blockIdx.x * blockDim.x + threadIdx.x; // < 3*NB*NT*NC
  const int per = NB * NT * NC;
  int isc  = (int)(idx / per);
  int rem  = (int)(idx % per);
  int bb   = rem / (NT * NC);
  int rem2 = rem % (NT * NC);
  int t    = rem2 / NC;
  int c    = rem2 % NC;
  const float* ap = (isc == 0) ? a0 : ((isc == 1) ? a1 : a2);
  const float* vp = (isc == 0) ? v0 : ((isc == 1) ? v1 : v2);
  float av = ap[rem];
  float vv = vp[rem];
  Abf[idx] = f2bf(av);
  Vbf[idx] = f2bf(vv);
  Abft[((size_t)(isc * NB + bb) * NC + c) * NT + t] = f2bf(av);
}

// ============= kernel 2: W[d][c] -> Wt[d][c] bf16 copy ============================
__global__ void k_convert_w(const float* __restrict__ W, unsigned short* __restrict__ Wt) {
  size_t idx = (size_t)blockIdx.x * blockDim.x + threadIdx.x; // < NC*NC
  int c = (int)(idx / NC);
  int d = (int)(idx % NC);
  Wt[idx] = f2bf(W[idx]);
}

// ============= kernel 3: WV = V @ W^T + b (bf16 WMMA, bf16 out) ==============
__global__ void k_linear(const unsigned short* __restrict__ Vbf,
                         const unsigned short* __restrict__ Wt,
                         const float* __restrict__ bias,
                         unsigned short* __restrict__ WVbf) {
  int wave = threadIdx.x >> 5;
  int lane = threadIdx.x & 31;
  int wid  = blockIdx.x * (blockDim.x >> 5) + wave;
  int tm = wid >> 5;  // 32 N-tiles per row block (512/16)
  int tn = wid & 31;
  int r = lane & 15;
  int h = lane >> 4;

  v8f acc = vzero();
  const unsigned short* arow = Vbf + (size_t)(tm * 16 + r) * NC;
#pragma unroll 4
  for (int kk = 0; kk < 16; ++kk) {
    int c0 = kk * 32;
    BFrag a, bf;
    a.q[0]  = *(const uint4*)(arow + c0 + h * 8);
    a.q[1]  = *(const uint4*)(arow + c0 + 16 + h * 8);
    const unsigned short* bp = Wt + (size_t)(tn * 16 + r) * NC + c0;
    bf.q[0] = *(const uint4*)(bp + h * 8);
    bf.q[1] = *(const uint4*)(bp + 16 + h * 8);
    acc = wmma_bf16(a.v, bf.v, acc);
  }
  float bb = bias[tn * 16 + r];
#pragma unroll
  for (int e = 0; e < 8; ++e) {
    int m = e + h * 8;
    WVbf[(size_t)(tm * 16 + m) * NC + tn * 16 + r] = f2bf(acc[e] + bb);
  }
}

// ============= kernel 4: scores + softmax + fv, and P^T to global ============
// grid: 72 combos * 16 t-blocks; block: 128 threads (4 waves, 16 t-rows each)
// dyn LDS: 4 waves * 16 rows * 4128 B = 264192 B
__global__ void k_attn_fv(const unsigned short* __restrict__ Abf,
                          const unsigned short* __restrict__ Abft,
                          const unsigned short* __restrict__ WVbf,
                          unsigned short* __restrict__ Pt,
                          float* __restrict__ out) {
  extern __shared__ char smem[];
  constexpr int ROWB = 4128; // (1024+8)*4 bytes, 16B aligned pitch
  int wave = threadIdx.x >> 5;
  int lane = threadIdx.x & 31;
  int combo = blockIdx.x >> 4;  // (i*3+j)*8 + b
  int tblk  = blockIdx.x & 15;
  int i = combo / 24;
  int j = (combo / 8) % 3;
  int b = combo & 7;
  int t0 = tblk * 64 + wave * 16;
  char* wbase = smem + (size_t)wave * 16 * ROWB;
  int r = lane & 15;
  int h = lane >> 4;

  // ---- preload Q fragments (16 rows x 512 d) in A layout ----
  BFrag qf[16];
  {
    const unsigned short* qrow = WVbf + (size_t)((i * NB + b) * NT + t0 + r) * NC;
#pragma unroll
    for (int kk = 0; kk < 16; ++kk) {
      qf[kk].q[0] = *(const uint4*)(qrow + kk * 32 + h * 8);
      qf[kk].q[1] = *(const uint4*)(qrow + kk * 32 + 16 + h * 8);
    }
  }

  // ---- phase A: scores -> LDS (f32) ----
  const unsigned short* KT = Abft + (size_t)(j * NB + b) * NC * NT; // [d][s]
  for (int st = 0; st < 64; ++st) {
    int s0 = st * 16;
    v8f acc = vzero();
#pragma unroll 4
    for (int kk = 0; kk < 16; ++kk) {
      BFrag kb;
      const unsigned short* kp = Abf + ((size_t)(j * NB + b) * NT + s0 + r) * NC + kk * 32;
      kb.q[0] = *(const uint4*)(kp + h * 8);
      kb.q[1] = *(const uint4*)(kp + 16 + h * 8);
      acc = wmma_bf16(qf[kk].v, kb.v, acc);
    }
#pragma unroll
    for (int e = 0; e < 8; ++e) {
      int m = e + h * 8;
      *(float*)(wbase + m * ROWB + (s0 + r) * 4) = acc[e] * SCALE;
    }
  }

  // ---- phase B: softmax stats (lane pair (r, r+16) shares row r) ----
  const float* myrow = (const float*)(wbase + r * ROWB);
  int sbase = h * 512;
  float mx = -1e30f;
  for (int k4 = 0; k4 < 128; ++k4) {
    float4 v = *(const float4*)(myrow + sbase + k4 * 4);
    mx = fmaxf(mx, fmaxf(fmaxf(v.x, v.y), fmaxf(v.z, v.w)));
  }
  mx = fmaxf(mx, __shfl_xor(mx, 16));
  float lsum = 0.0f;
  for (int k4 = 0; k4 < 128; ++k4) {
    float4 v = *(const float4*)(myrow + sbase + k4 * 4);
    lsum += __expf(v.x - mx) + __expf(v.y - mx) + __expf(v.z - mx) + __expf(v.w - mx);
  }
  lsum += __shfl_xor(lsum, 16);
  float invl = 1.0f / lsum;

  // ---- phase B2: normalized P (bf16) in place, each lane compacts own half ----
  for (int k4 = 0; k4 < 128; ++k4) {
    int s = sbase + k4 * 4;
    float4 v = *(const float4*)(myrow + s);
    unsigned short p0 = f2bf(__expf(v.x - mx) * invl);
    unsigned short p1 = f2bf(__expf(v.y - mx) * invl);
    unsigned short p2 = f2bf(__expf(v.z - mx) * invl);
    unsigned short p3 = f2bf(__expf(v.w - mx) * invl);
    uint2 w;
    w.x = (unsigned)p0 | ((unsigned)p1 << 16);
    w.y = (unsigned)p2 | ((unsigned)p3 << 16);
    *(uint2*)(wbase + r * ROWB + pOff(s)) = w;
  }

  // ---- phase C: fv = P @ A_j, c-chunked (accum stays in 8 VGPRs) ----
  const unsigned short* Arow = Abf + (size_t)(j * NB + b) * NT * NC;
  size_t ob = ((size_t)((i * 3 + j) * NB + b) * NT + t0) * NC;
  for (int cc = 0; cc < 32; ++cc) {
    v8f acc = vzero();
#pragma unroll 4
    for (int kt = 0; kt < 32; ++kt) {
      int kb = kt * 32;
      BFrag pa, ab;
      pa.q[0] = *(const uint4*)(wbase + r * ROWB + pOff(kb + h * 8));
      pa.q[1] = *(const uint4*)(wbase + r * ROWB + pOff(kb + 16 + h * 8));
      const unsigned short* ap = Abft + ((size_t)(j * NB + b) * NC + cc * 16 + r) * NT + kb;
      ab.q[0] = *(const uint4*)(ap + h * 8);
      ab.q[1] = *(const uint4*)(ap + 16 + h * 8);
      acc = wmma_bf16(pa.v, ab.v, acc);
    }
#pragma unroll
    for (int e = 0; e < 8; ++e)
      out[ob + (size_t)(e + h * 8) * NC + cc * 16 + r] = acc[e];
  }

  // ---- phase D: stream P^T (bf16) to global workspace for the fa pass ----
  unsigned short* PtC = Pt + (size_t)combo * NT * NT;
  for (int s0 = 0; s0 < NT; s0 += 32) {
    int s = s0 + lane;
    unsigned short vals[16];
#pragma unroll
    for (int t = 0; t < 16; ++t)
      vals[t] = *(const unsigned short*)(wbase + t * ROWB + pOff(s));
    uint4 w0, w1;
    w0.x = (unsigned)vals[0]  | ((unsigned)vals[1]  << 16);
    w0.y = (unsigned)vals[2]  | ((unsigned)vals[3]  << 16);
    w0.z = (unsigned)vals[4]  | ((unsigned)vals[5]  << 16);
    w0.w = (unsigned)vals[6]  | ((unsigned)vals[7]  << 16);
    w1.x = (unsigned)vals[8]  | ((unsigned)vals[9]  << 16);
    w1.y = (unsigned)vals[10] | ((unsigned)vals[11] << 16);
    w1.z = (unsigned)vals[12] | ((unsigned)vals[13] << 16);
    w1.w = (unsigned)vals[14] | ((unsigned)vals[15] << 16);
    uint4* dst = (uint4*)(PtC + (size_t)s * NT + t0);
    dst[0] = w0;
    dst[1] = w1;
  }
}

// ============= kernel 5: fa = P^T @ V with async-to-LDS V staging ============
// block: 128 threads (4 waves). LDS: double-buffered V strip, 2 * 32KB.
// Strip cc = V[i,b][t=0..1023][cc*16 .. cc*16+16) staged by ASYNC DMA, shared
// by all 4 waves; next strip's DMA overlaps the current 32-WMMA chunk.
__global__ void k_fa(const unsigned short* __restrict__ Pt,
                     const unsigned short* __restrict__ Vbf,
                     float* __restrict__ out) {
  extern __shared__ char smem[];
  int tid  = threadIdx.x;
  int wave = tid >> 5;
  int lane = tid & 31;
  int combo = blockIdx.x >> 4;
  int sblk  = blockIdx.x & 15;
  int i = combo / 24;
  int j = (combo / 8) % 3;
  int b = combo & 7;
  int s0 = sblk * 64 + wave * 16;
  int r = lane & 15;
  int h = lane >> 4;

  const unsigned short* Prow = Pt + (size_t)combo * NT * NT + (size_t)(s0 + r) * NT;
  const unsigned short* Vrow = Vbf + (size_t)(i * NB + b) * NT * NC;
  size_t ob = (size_t)3 * 3 * NB * NT * NC  // fa half of the output
            + ((size_t)((j * 3 + i) * NB + b) * NT + s0) * NC;

  // issue one strip: 1024 rows x 32B; 128 threads x 8 rows x 2 b128
  auto issue_strip = [&](int cc, int buf) {
    unsigned base = (unsigned)buf * 32768u;
    const unsigned short* g0 = Vrow + cc * 16;
#pragma unroll
    for (int k = 0; k < 8; ++k) {
      int row = tid + k * 128;
      unsigned lds_off = base + (unsigned)row * 32u;
      const void* ga = (const void*)(g0 + (size_t)row * NC);
      async_g2l_b128(lds_off, ga);
      async_g2l_b128_o16(lds_off, ga);  // offset:16 advances global AND LDS addr
    }
  };

  issue_strip(0, 0);
  for (int cc = 0; cc < 32; ++cc) {
    wait_async0();        // my wave's share of strip cc has landed in LDS
    __syncthreads();      // everyone's share landed
    if (cc + 1 < 32) issue_strip(cc + 1, (cc + 1) & 1);  // overlap with compute

    const char* sbase = smem + (cc & 1) * 32768;
    v8f acc = vzero();
#pragma unroll 4
    for (int kt = 0; kt < 32; ++kt) {
      int kb = kt * 32;
      BFrag pa, vb;
      pa.q[0] = *(const uint4*)(Prow + kb + h * 8);
      pa.q[1] = *(const uint4*)(Prow + kb + 16 + h * 8);
#pragma unroll
      for (int e = 0; e < 16; ++e) {
        int kr = kb + ((e < 8) ? h * 8 + e : 16 + h * 8 + (e - 8));
        ((unsigned short*)&vb)[e] = *(const unsigned short*)(sbase + kr * 32 + r * 2);
      }
      acc = wmma_bf16(pa.v, vb.v, acc);
    }
#pragma unroll
    for (int e = 0; e < 8; ++e)
      out[ob + (size_t)(e + h * 8) * NC + cc * 16 + r] = acc[e];
    __syncthreads();      // all waves done reading this buffer before reuse
  }
}

// ================================ launcher ===================================
extern "C" void kernel_launch(void* const* d_in, const int* in_sizes, int n_in,
                              void* d_out, int out_size, void* d_ws, size_t ws_size,
                              hipStream_t stream) {
  const float* a0 = (const float*)d_in[0];
  const float* a1 = (const float*)d_in[1];
  const float* a2 = (const float*)d_in[2];
  const float* v0 = (const float*)d_in[3];
  const float* v1 = (const float*)d_in[4];
  const float* v2 = (const float*)d_in[5];
  const float* W  = (const float*)d_in[6];
  const float* bv = (const float*)d_in[7];
  float* out = (float*)d_out;

  char* ws = (char*)d_ws;
  const size_t nAV = (size_t)3 * NB * NT * NC; // 12,582,912 elems
  unsigned short* Abf  = (unsigned short*)ws;                     ws += nAV * 2;
  unsigned short* Abft = (unsigned short*)ws;                     ws += nAV * 2;
  unsigned short* Vbf  = (unsigned short*)ws;                     ws += nAV * 2;
  unsigned short* WVbf = (unsigned short*)ws;                     ws += nAV * 2;
  unsigned short* Wt   = (unsigned short*)ws;                     ws += (size_t)NC * NC * 2;
  unsigned short* Pt   = (unsigned short*)ws;                     // 9*B*T*T bf16 = 151 MB

  // 1) fp32 -> bf16 (+ A transpose)
  k_convert<<<(unsigned)(nAV / 256), 256, 0, stream>>>(a0, a1, a2, v0, v1, v2,
                                                       Abf, Abft, Vbf);
  // 2) W -> bf16 copy (same [d][c] layout)
  k_convert_w<<<(NC * NC) / 256, 256, 0, stream>>>(W, Wt);
  // 3) WV = V @ W^T + b   (24576x512x512, 16x16 tile per wave)
  k_linear<<<6144, 256, 0, stream>>>(Vbf, Wt, bv, WVbf);
  // 4) scores + softmax + fv, P^T to workspace (72 combos x 16 t-blocks)
  k_attn_fv<<<72 * 16, 128, 4 * 16 * 4128, stream>>>(Abf, Abft, WVbf, Pt, out);
  // 5) fa = P^T @ V with double-buffered async V staging
  k_fa<<<72 * 16, 128, 2 * 32768, stream>>>(Pt, Vbf, out);
}
